// MLPAttention_64896955843210
// MI455X (gfx1250) — matrix operand order repaired and checked by reading, not compiled, not run on hardware
//
#include <hip/hip_runtime.h>

#define T_DIM 128
#define S_DIM 256
#define B_DIM 16
#define H_DIM 256
#define A_DIM 256

typedef __attribute__((ext_vector_type(16))) __bf16 v16bf;
typedef __attribute__((ext_vector_type(8)))  float  v8f;

// ---------------------------------------------------------------------------
// Fragment builders, direct from global memory (wave32, 16x16x32 bf16 WMMA).
// This code uses K-contiguous per-lane element order (k = half*16 + s, same for A and B):
//  A (16x32): lane m=lane%16, half=lane/16:
//     f[0..7]  = A[m][half*16 + 0..7]
//     f[8..15] = A[m][half*16 + 8 + 0..7]
//  B (32x16) from transposed weights Bt[n][k]: lane n=lane%16:
//     f[0..15] = Bt[n][half*16 + 0..15]
// So each fragment = 4 x global_load_b128 + packed f32->bf16 cvt. The 32
// lanes cover each 16x32 tile exactly once (no redundant traffic).
// ---------------------------------------------------------------------------
__device__ __forceinline__ void put8(v16bf& f, int base, float4 a, float4 b) {
  f[base + 0] = (__bf16)a.x; f[base + 1] = (__bf16)a.y;
  f[base + 2] = (__bf16)a.z; f[base + 3] = (__bf16)a.w;
  f[base + 4] = (__bf16)b.x; f[base + 5] = (__bf16)b.y;
  f[base + 6] = (__bf16)b.z; f[base + 7] = (__bf16)b.w;
}

__device__ __forceinline__ v16bf frag_a_g(const float* __restrict__ A /*tile origin @k0*/,
                                          int lda, int lane) {
  const int m = lane & 15, half = lane >> 4;
  const float* p0 = A + (size_t)m * lda + half * 16;
  const float* p1 = p0 + 8;
  v16bf f;
  put8(f, 0, ((const float4*)p0)[0], ((const float4*)p0)[1]);
  put8(f, 8, ((const float4*)p1)[0], ((const float4*)p1)[1]);
  return f;
}

__device__ __forceinline__ v16bf frag_b_g_nk(const float* __restrict__ Bt /*[n][k] @(n0,k0)*/,
                                             int ldb, int lane) {
  const int n = lane & 15, half = lane >> 4;
  const float4* p = (const float4*)(Bt + (size_t)n * ldb + half * 16);
  v16bf f;
  put8(f, 0, p[0], p[1]);
  put8(f, 8, p[2], p[3]);
  return f;
}

__device__ __forceinline__ v16bf frag_b_g_kn(const float* __restrict__ Bm /*[k][n] @(k0,n0)*/,
                                             int ldk, int lane) {
  const int n = lane & 15, half = lane >> 4;
  v16bf f;
#pragma unroll
  for (int i = 0; i < 16; ++i) f[i] = (__bf16)Bm[(size_t)(half * 16 + i) * ldk + n];
  return f;
}

__device__ __forceinline__ v8f wmma_bf16(v16bf a, v16bf b, v8f c) {
  // (neg_a, A, neg_b, B, c_mod, C, reuse_a, reuse_b)
  return __builtin_amdgcn_wmma_f32_16x16x32_bf16(false, a, false, b, (short)0, c,
                                                 false, false);
}

// ---------------------------------------------------------------------------
// qp = q @ Wq^T + bq ; rows r = b*T+t, q stored time-major (T,B,H).
// One wave computes a 16x64 tile (4 WMMA accumulators sharing one A frag).
// ---------------------------------------------------------------------------
__global__ __launch_bounds__(128) void qp_gemm(const float* __restrict__ q,
                                               const float* __restrict__ Wq,
                                               const float* __restrict__ bq,
                                               float* __restrict__ qp) {
  const int wid = blockIdx.x * 4 + (threadIdx.x >> 5);  // 512 waves
  const int lane = threadIdx.x & 31;
  const int tm = wid >> 2;   // 128 M-tiles of 16 rows
  const int tn4 = wid & 3;   // 4 N-groups of 64 cols
  const int r0 = tm * 16;
  const int b = r0 / T_DIM, t0 = r0 % T_DIM;  // b constant within 16-row tile
  const float* Abase = q + ((size_t)t0 * B_DIM + b) * H_DIM;
  const int lda = B_DIM * H_DIM;
  v8f acc[4] = {};
#pragma unroll
  for (int k0 = 0; k0 < H_DIM; k0 += 32) {
    const v16bf a = frag_a_g(Abase + k0, lda, lane);
#pragma unroll
    for (int u = 0; u < 4; ++u) {
      const int n0 = tn4 * 64 + u * 16;
      const v16bf bf = frag_b_g_nk(Wq + (size_t)n0 * H_DIM + k0, H_DIM, lane);
      acc[u] = wmma_bf16(a, bf, acc[u]);
    }
  }
  const int n = lane & 15, mb = (lane >> 4) * 8;
#pragma unroll
  for (int u = 0; u < 4; ++u) {
    const int h = tn4 * 64 + u * 16 + n;
    const float bias = bq[h];
#pragma unroll
    for (int v = 0; v < 8; ++v)
      qp[(size_t)(r0 + mb + v) * H_DIM + h] = acc[u][v] + bias;
  }
}

// ---------------------------------------------------------------------------
// ap = a @ Wa^T ; rows r = b*S+s, annotations stored (S,B,A).
// ---------------------------------------------------------------------------
__global__ __launch_bounds__(128) void ap_gemm(const float* __restrict__ ann,
                                               const float* __restrict__ Wa,
                                               float* __restrict__ ap) {
  const int wid = blockIdx.x * 4 + (threadIdx.x >> 5);  // 1024 waves
  const int lane = threadIdx.x & 31;
  const int tm = wid >> 2;   // 256 M-tiles
  const int tn4 = wid & 3;
  const int r0 = tm * 16;
  const int b = r0 / S_DIM, s0 = r0 % S_DIM;
  const float* Abase = ann + ((size_t)s0 * B_DIM + b) * A_DIM;
  const int lda = B_DIM * A_DIM;
  v8f acc[4] = {};
#pragma unroll
  for (int k0 = 0; k0 < A_DIM; k0 += 32) {
    const v16bf a = frag_a_g(Abase + k0, lda, lane);
#pragma unroll
    for (int u = 0; u < 4; ++u) {
      const int n0 = tn4 * 64 + u * 16;
      const v16bf bf = frag_b_g_nk(Wa + (size_t)n0 * A_DIM + k0, A_DIM, lane);
      acc[u] = wmma_bf16(a, bf, acc[u]);
    }
  }
  const int n = lane & 15, mb = (lane >> 4) * 8;
#pragma unroll
  for (int u = 0; u < 4; ++u) {
#pragma unroll
    for (int v = 0; v < 8; ++v)
      ap[(size_t)(r0 + mb + v) * H_DIM + tn4 * 64 + u * 16 + n] = acc[u][v];
  }
}

// ---------------------------------------------------------------------------
// logits[s] = sum_h tanh(qp[b,t,h] + ap[b,s,h]) * v_w[h]; softmax over s.
// One block per (b,t). Each lane owns h-slice [lane*8, lane*8+8) in registers.
// Dominant cost of the whole problem: B*T*S*H = 134M v_tanh_f32 (TRANS pipe).
// ---------------------------------------------------------------------------
__global__ __launch_bounds__(256) void score_kernel(const float* __restrict__ qp,
                                                    const float* __restrict__ ap,
                                                    const float* __restrict__ v_w,
                                                    float* __restrict__ scores_out) {
  const int bt = blockIdx.x;  // b*T + t
  const int b = bt / T_DIM, t = bt % T_DIM;
  const int wave = threadIdx.x >> 5, lane = threadIdx.x & 31;
  __shared__ float sh_logit[S_DIM];
  __shared__ float red[256];

  // per-lane register slice of qp row and v_w
  const float4* qv = (const float4*)(qp + (size_t)bt * H_DIM + lane * 8);
  const float4* wv = (const float4*)(v_w + lane * 8);
  const float4 q0 = qv[0], q1 = qv[1];
  const float4 w0 = wv[0], w1 = wv[1];

  for (int s = wave; s < S_DIM; s += 8) {
    const float4* av = (const float4*)(ap + ((size_t)b * S_DIM + s) * H_DIM + lane * 8);
    const float4 a0 = av[0], a1 = av[1];
    float part = tanhf(q0.x + a0.x) * w0.x + tanhf(q0.y + a0.y) * w0.y +
                 tanhf(q0.z + a0.z) * w0.z + tanhf(q0.w + a0.w) * w0.w +
                 tanhf(q1.x + a1.x) * w1.x + tanhf(q1.y + a1.y) * w1.y +
                 tanhf(q1.z + a1.z) * w1.z + tanhf(q1.w + a1.w) * w1.w;
#pragma unroll
    for (int off = 16; off > 0; off >>= 1) part += __shfl_down(part, off, 32);
    if (lane == 0) sh_logit[s] = part;
  }
  __syncthreads();
  const float x = sh_logit[threadIdx.x];
  red[threadIdx.x] = x;
  __syncthreads();
  for (int off = 128; off > 0; off >>= 1) {
    if (threadIdx.x < off) red[threadIdx.x] = fmaxf(red[threadIdx.x], red[threadIdx.x + off]);
    __syncthreads();
  }
  const float mx = red[0];
  __syncthreads();
  const float e = __expf(x - mx);
  red[threadIdx.x] = e;
  __syncthreads();
  for (int off = 128; off > 0; off >>= 1) {
    if (threadIdx.x < off) red[threadIdx.x] += red[threadIdx.x + off];
    __syncthreads();
  }
  scores_out[((size_t)t * B_DIM + b) * S_DIM + threadIdx.x] = e / red[0];
}

// ---------------------------------------------------------------------------
// ctx[b,t,:] = scores[b,t,:] @ a[b]  (grid.y = b). B matrix is row-major KxN.
// ---------------------------------------------------------------------------
__global__ __launch_bounds__(128) void ctx_gemm(const float* __restrict__ scores,
                                                const float* __restrict__ ann,
                                                float* __restrict__ ctx) {
  const int wid = blockIdx.x * 4 + (threadIdx.x >> 5);  // 32 waves per batch
  const int lane = threadIdx.x & 31;
  const int b = blockIdx.y;
  const int tm = wid >> 2;   // 8 T-tiles
  const int tn4 = wid & 3;   // 4 N-groups of 64
  const int t0 = tm * 16;
  const float* Abase = scores + ((size_t)t0 * B_DIM + b) * S_DIM;
  const int lda = B_DIM * S_DIM;
  const int ldk = B_DIM * A_DIM;
  v8f acc[4] = {};
#pragma unroll
  for (int k0 = 0; k0 < S_DIM; k0 += 32) {
    const v16bf a = frag_a_g(Abase + k0, lda, lane);
#pragma unroll
    for (int u = 0; u < 4; ++u) {
      const int n0 = tn4 * 64 + u * 16;
      const v16bf bf =
          frag_b_g_kn(ann + ((size_t)k0 * B_DIM + b) * A_DIM + n0, ldk, lane);
      acc[u] = wmma_bf16(a, bf, acc[u]);
    }
  }
  const int n = lane & 15, mb = (lane >> 4) * 8;
#pragma unroll
  for (int u = 0; u < 4; ++u) {
#pragma unroll
    for (int v = 0; v < 8; ++v)
      ctx[((size_t)b * T_DIM + t0 + mb + v) * A_DIM + tn4 * 64 + u * 16 + n] = acc[u][v];
  }
}

// ---------------------------------------------------------------------------
// att = tanh([q, ctx] @ W_att^T + b_att), K = 512 (concat folded into K-loop).
// ---------------------------------------------------------------------------
__global__ __launch_bounds__(128) void att_gemm(const float* __restrict__ q,
                                                const float* __restrict__ ctx,
                                                const float* __restrict__ Watt,
                                                const float* __restrict__ batt,
                                                float* __restrict__ att) {
  const int wid = blockIdx.x * 4 + (threadIdx.x >> 5);  // 512 waves
  const int lane = threadIdx.x & 31;
  const int tm = wid >> 2;
  const int tn4 = wid & 3;
  const int r0 = tm * 16;
  const int b = r0 / T_DIM, t0 = r0 % T_DIM;
  const float* Aq = q + ((size_t)t0 * B_DIM + b) * H_DIM;   // K in [0,256)
  const float* Ac = ctx + (size_t)r0 * A_DIM;               // K in [256,512)
  const int KTOT = H_DIM + A_DIM;
  v8f acc[4] = {};
#pragma unroll
  for (int k0 = 0; k0 < KTOT; k0 += 32) {
    const v16bf a = (k0 < H_DIM)
                        ? frag_a_g(Aq + k0, B_DIM * H_DIM, lane)
                        : frag_a_g(Ac + (k0 - H_DIM), A_DIM, lane);
#pragma unroll
    for (int u = 0; u < 4; ++u) {
      const int n0 = tn4 * 64 + u * 16;
      const v16bf bf = frag_b_g_nk(Watt + (size_t)n0 * KTOT + k0, KTOT, lane);
      acc[u] = wmma_bf16(a, bf, acc[u]);
    }
  }
  const int n = lane & 15, mb = (lane >> 4) * 8;
#pragma unroll
  for (int u = 0; u < 4; ++u) {
    const int h = tn4 * 64 + u * 16 + n;
    const float bias = batt[h];
#pragma unroll
    for (int v = 0; v < 8; ++v) {
      const int t = t0 + mb + v;
      att[((size_t)t * B_DIM + b) * H_DIM + h] = tanhf(acc[u][v] + bias);
    }
  }
}

// ---------------------------------------------------------------------------
extern "C" void kernel_launch(void* const* d_in, const int* in_sizes, int n_in,
                              void* d_out, int out_size, void* d_ws, size_t ws_size,
                              hipStream_t stream) {
  const float* queries     = (const float*)d_in[0];  // (T,B,H)
  const float* annotations = (const float*)d_in[1];  // (S,B,A)
  const float* Wq          = (const float*)d_in[2];  // (H,H)
  const float* bq          = (const float*)d_in[3];  // (H)
  const float* Wa          = (const float*)d_in[4];  // (H,A)
  const float* v_w         = (const float*)d_in[5];  // (H)
  const float* W_att       = (const float*)d_in[6];  // (H, H+A)
  const float* b_att       = (const float*)d_in[7];  // (H)

  float* att_out    = (float*)d_out;                            // T*B*H
  float* scores_out = att_out + (size_t)T_DIM * B_DIM * H_DIM;  // T*B*S

  float* qp_ws  = (float*)d_ws;                                 // 2048*256
  float* ap_ws  = qp_ws + (size_t)B_DIM * T_DIM * H_DIM;        // 4096*256
  float* ctx_ws = ap_ws + (size_t)B_DIM * S_DIM * H_DIM;        // 2048*256

  // 1) qp = q@Wq^T + bq : 512 waves (16x64 tile each)
  qp_gemm<<<128, 128, 0, stream>>>(queries, Wq, bq, qp_ws);
  // 2) ap = a@Wa^T : 1024 waves
  ap_gemm<<<256, 128, 0, stream>>>(annotations, Wa, ap_ws);
  // 3) tanh-dot logits + softmax -> scores (T,B,S) in d_out
  score_kernel<<<B_DIM * T_DIM, 256, 0, stream>>>(qp_ws, ap_ws, v_w, scores_out);
  // 4) ctx = scores @ a (batched over b)
  ctx_gemm<<<dim3(8, B_DIM), 128, 0, stream>>>(scores_out, annotations, ctx_ws);
  // 5) att = tanh([q,ctx]@W_att^T + b_att) -> (T,B,H) in d_out
  att_gemm<<<128, 128, 0, stream>>>(queries, ctx_ws, W_att, b_att, att_out);
}
